// FlashBlockwiseAttentionLayer_32762010534198
// MI455X (gfx1250) — compile-verified, matrix operand order checked by reading, not run on hardware
//
#include <hip/hip_runtime.h>

// ---------------------------------------------------------------------------
// Problem constants (reference: B=4, S=1024, HID=2048, H=16, D=128, CLEN=2048)
// ---------------------------------------------------------------------------
#define BATCH 4
#define SEQ   1024
#define HID   2048
#define NH    16
#define HD    128
#define CLEN  2048
#define KVLEN (CLEN + SEQ)   // 3072

typedef __attribute__((ext_vector_type(16))) __bf16 v16bf;
typedef __attribute__((ext_vector_type(8)))  __bf16 v8bf;
typedef __attribute__((ext_vector_type(8)))  float  v8f;

union Frag { v16bf v; v8bf h[2]; };

// ---------------------------------------------------------------------------
// CDNA5 async memory->LDS copy (ASYNCcnt-tracked). IOFFSET is added to both
// the LDS and global addresses, so one base pair covers a 32B slice.
// ---------------------------------------------------------------------------
__device__ __forceinline__ void async_copy32(unsigned lds_off, const __bf16* g) {
    asm volatile(
        "global_load_async_to_lds_b128 %0, %1, off\n\t"
        "global_load_async_to_lds_b128 %0, %1, off offset:16"
        :: "v"(lds_off), "v"(g) : "memory");
}

// ---------------------------------------------------------------------------
// Elementwise f32 -> bf16
// ---------------------------------------------------------------------------
__global__ void cvt_f32_bf16(const float* __restrict__ in, __bf16* __restrict__ out, int n) {
    int i = blockIdx.x * 256 + threadIdx.x;
    if (i < n) out[i] = (__bf16)in[i];
}

// cached_v [CLEN,H,D] f32 -> Vt cache [H,D,CLEN] bf16 (PV B-fragments contiguous)
__global__ void cvt_vcache_t(const float* __restrict__ v, __bf16* __restrict__ vt) {
    int i = blockIdx.x * 256 + threadIdx.x;
    if (i < CLEN * NH * HD) {
        int d = i & (HD - 1);
        int h = (i >> 7) & (NH - 1);
        int c = i >> 11;
        vt[((size_t)(h * HD + d)) * CLEN + c] = (__bf16)v[i];
    }
}

// ---------------------------------------------------------------------------
// Tiled WMMA GEMM:  C[M,N] = A[M,K] * W[N,K]^T + bias   (K = 2048, M = 4096)
// All operands bf16. Tiles staged with GLOBAL_LOAD_ASYNC_TO_LDS_B128 into a
// double-buffered LDS pipeline (issue next tile, s_wait_asynccnt on current).
//   EPI=0: write f32 to Cout.   EPI=1: scatter QKV -> Q/Knew/VtNew (bf16).
// Block: 256 threads (8 waves), 128x128 output tile, BK=32.
// Wave grid 2x4; each wave owns 64x32 = 4x2 WMMA accumulators.
// ---------------------------------------------------------------------------
template<int EPI>
__global__ __launch_bounds__(256)
void gemm_wmma(const __bf16* __restrict__ A, const __bf16* __restrict__ W,
               const float* __restrict__ bias, float* __restrict__ Cout,
               __bf16* __restrict__ qOut, __bf16* __restrict__ kOut,
               __bf16* __restrict__ vtOut, int N)
{
    constexpr int K = HID;
    __shared__ __bf16 As[2][128][32];
    __shared__ __bf16 Ws[2][128][32];

    const int tid   = threadIdx.x;
    const int lane  = tid & 31;
    const int wid   = tid >> 5;
    const int waveM = wid >> 2;          // 0..1
    const int waveN = wid & 3;           // 0..3
    const int hl    = lane >> 4;         // 0/1 lane-half
    const int nl    = lane & 15;
    const int mBlock = blockIdx.y * 128;
    const int nBlock = blockIdx.x * 128;
    const int srow = tid >> 1;           // staging row 0..127
    const int skh  = (tid & 1) * 16;     // staging k-offset 0/16

    const __bf16* aBase = A + (size_t)(mBlock + srow) * K + skh;
    const __bf16* wBase = W + (size_t)(nBlock + srow) * K + skh;

    v8f acc[4][2];
    for (int mt = 0; mt < 4; ++mt)
        for (int nt = 0; nt < 2; ++nt)
            for (int i = 0; i < 8; ++i) acc[mt][nt][i] = 0.0f;

    // prologue: async-stage tile 0 into buffer 0
    async_copy32((unsigned)(uintptr_t)&As[0][srow][skh], aBase);
    async_copy32((unsigned)(uintptr_t)&Ws[0][srow][skh], wBase);

    for (int kb = 0; kb < K; kb += 32) {
        const int p = (kb >> 5) & 1;
        if (kb + 32 < K) {
            // issue next tile into the other buffer, then wait for current
            async_copy32((unsigned)(uintptr_t)&As[1 - p][srow][skh], aBase + kb + 32);
            async_copy32((unsigned)(uintptr_t)&Ws[1 - p][srow][skh], wBase + kb + 32);
            asm volatile("s_wait_asynccnt 0x4" ::: "memory");
        } else {
            asm volatile("s_wait_asynccnt 0x0" ::: "memory");
        }
        __syncthreads();

        // ---- build fragments from LDS (ISA A/B layouts: two 16B chunks/lane) ----
        Frag af[4], wf[2];
        for (int mt = 0; mt < 4; ++mt) {
            const int ar = waveM * 64 + mt * 16 + nl;
            af[mt].h[0] = *(const v8bf*)&As[p][ar][hl * 8];
            af[mt].h[1] = *(const v8bf*)&As[p][ar][16 + hl * 8];
        }
        for (int nt = 0; nt < 2; ++nt) {
            const int wr = waveN * 32 + nt * 16 + nl;
            wf[nt].h[0] = *(const v8bf*)&Ws[p][wr][hl * 8];
            wf[nt].h[1] = *(const v8bf*)&Ws[p][wr][16 + hl * 8];
        }
        for (int mt = 0; mt < 4; ++mt)
            for (int nt = 0; nt < 2; ++nt)
                acc[mt][nt] = __builtin_amdgcn_wmma_f32_16x16x32_bf16(
                    false, af[mt].v, false, wf[nt].v, (short)0, acc[mt][nt], false, false);
        __syncthreads();   // all waves done reading buf[p] before it is refilled
    }

    // ---- epilogue ----
    for (int mt = 0; mt < 4; ++mt) {
        for (int nt = 0; nt < 2; ++nt) {
            for (int i = 0; i < 8; ++i) {
                const int grow = mBlock + waveM * 64 + mt * 16 + hl * 8 + i;  // 0..4095
                const int gcol = nBlock + waveN * 32 + nt * 16 + nl;          // 0..N-1
                const float val = acc[mt][nt][i] + bias[gcol];
                if constexpr (EPI == 0) {
                    Cout[(size_t)grow * N + gcol] = val;
                } else {
                    const int b = grow >> 10, s = grow & (SEQ - 1);
                    const int which = gcol >> 11, idx = gcol & (HID - 1);
                    const int h = idx >> 7, d = idx & (HD - 1);
                    if (which == 0)
                        qOut[(((size_t)(b * NH + h) * SEQ + s) << 7) + d] = (__bf16)val;
                    else if (which == 1)
                        kOut[(((size_t)(b * NH + h) * SEQ + s) << 7) + d] = (__bf16)val;
                    else
                        vtOut[(((size_t)(b * NH + h) << 7) + d) * SEQ + s] = (__bf16)val;
                }
            }
        }
    }
}

// ---------------------------------------------------------------------------
// Flash attention. Grid = B*H*(S/64) blocks of 128 threads (4 waves).
// Each wave: one 16-row Q tile, streams KV in 32-key blocks (KV lives in L2).
//   S^T tile (keys x q) via  K(A) x Q^T(B)  -> per-lane column softmax.
//   The S^T C-tile lane layout == P A-fragment lane layout, so P packs
//   in-register (no LDS transpose needed).  O += P x V via contiguous V^T cols.
// ---------------------------------------------------------------------------
__global__ __launch_bounds__(128)
void flash_attn(const __bf16* __restrict__ Q,  const __bf16* __restrict__ Kc,
                const __bf16* __restrict__ Kn, const __bf16* __restrict__ Vtc,
                const __bf16* __restrict__ Vtn, __bf16* __restrict__ attnOut)
{
    const int wid  = threadIdx.x >> 5;
    const int lane = threadIdx.x & 31;
    const int hl   = lane >> 4;
    const int nl   = lane & 15;

    const int bid = blockIdx.x;
    const int b   = bid >> 8;            // 256 blocks per batch
    const int h   = (bid >> 4) & 15;
    const int q0  = (bid & 15) * 64 + wid * 16;

    // Q fragments: B operand of S^T = K x Q^T; column nl = row (q0+nl) of Q.
    Frag qf[4];
    const __bf16* qrow = Q + ((size_t)(b * NH + h) * SEQ + q0 + nl) * HD;
    for (int dc = 0; dc < 4; ++dc) {
        qf[dc].h[0] = *(const v8bf*)(qrow + dc * 32 + hl * 8);
        qf[dc].h[1] = *(const v8bf*)(qrow + dc * 32 + 16 + hl * 8);
    }

    v8f o[8];
    for (int dt = 0; dt < 8; ++dt)
        for (int i = 0; i < 8; ++i) o[dt][i] = 0.0f;
    float mrun = -1e30f, lrun = 0.0f;
    const float scale = 0.08838834764831845f;   // 1/sqrt(128)

    for (int kk = 0; kk < KVLEN; kk += 32) {
        const bool inc = kk < CLEN;             // 2048 % 32 == 0: clean boundary
        // ---- scores S^T (two 16-key tiles) ----
        v8f st[2];
        for (int t = 0; t < 2; ++t) {
            for (int i = 0; i < 8; ++i) st[t][i] = 0.0f;
            const int key = kk + t * 16 + nl;
            const __bf16* krow = inc
                ? (Kc + ((size_t)key * NH + h) * HD)
                : (Kn + ((size_t)(b * NH + h) * SEQ + (key - CLEN)) * HD);
            for (int dc = 0; dc < 4; ++dc) {
                Frag kf;
                kf.h[0] = *(const v8bf*)(krow + dc * 32 + hl * 8);
                kf.h[1] = *(const v8bf*)(krow + dc * 32 + 16 + hl * 8);
                st[t] = __builtin_amdgcn_wmma_f32_16x16x32_bf16(
                    false, kf.v, false, qf[dc].v, (short)0, st[t], false, false);
            }
        }
        // ---- online softmax per q-column (lane-local + one shfl_xor 16) ----
        float bm = -1e30f;
        for (int t = 0; t < 2; ++t)
            for (int i = 0; i < 8; ++i) {
                st[t][i] *= scale;
                bm = fmaxf(bm, st[t][i]);
            }
        bm = fmaxf(bm, __shfl_xor(bm, 16, 32));
        const float mnew  = fmaxf(mrun, bm);
        const float alpha = __expf(mrun - mnew);
        float rsum = 0.0f;
        for (int t = 0; t < 2; ++t)
            for (int i = 0; i < 8; ++i) {
                st[t][i] = __expf(st[t][i] - mnew);
                rsum += st[t][i];
            }
        rsum += __shfl_xor(rsum, 16, 32);
        lrun = lrun * alpha + rsum;
        mrun = mnew;
        // rescale O accumulators: alpha for row m lives in lane m
        for (int i = 0; i < 8; ++i) {
            const float ai = __shfl(alpha, hl * 8 + i, 32);
            for (int dt = 0; dt < 8; ++dt) o[dt][i] *= ai;
        }
        // ---- P packs directly into an A-fragment (layout coincidence) ----
        Frag pf;
        for (int i = 0; i < 8; ++i) {
            pf.h[0][i] = (__bf16)st[0][i];
            pf.h[1][i] = (__bf16)st[1][i];
        }
        // ---- O += P x V (V^T columns are contiguous in key) ----
        for (int dt = 0; dt < 8; ++dt) {
            const int dcol = dt * 16 + nl;
            const __bf16* vcol = inc
                ? (Vtc + ((size_t)(h * HD + dcol)) * CLEN + kk)
                : (Vtn + ((size_t)((b * NH + h) * HD + dcol)) * SEQ + (kk - CLEN));
            Frag vf;
            vf.h[0] = *(const v8bf*)(vcol + hl * 8);
            vf.h[1] = *(const v8bf*)(vcol + 16 + hl * 8);
            o[dt] = __builtin_amdgcn_wmma_f32_16x16x32_bf16(
                false, pf.v, false, vf.v, (short)0, o[dt], false, false);
        }
    }

    // ---- normalize and write attn output [B, S, H*D] as bf16 for proj GEMM ----
    for (int i = 0; i < 8; ++i) {
        const float li  = __shfl(lrun, hl * 8 + i, 32);
        const float inv = 1.0f / li;
        const int s = q0 + hl * 8 + i;
        __bf16* dst = attnOut + ((size_t)(b * SEQ + s)) * HID + h * HD;
        for (int dt = 0; dt < 8; ++dt)
            dst[dt * 16 + nl] = (__bf16)(o[dt][i] * inv);
    }
}

// ---------------------------------------------------------------------------
// Launch
// ---------------------------------------------------------------------------
extern "C" void kernel_launch(void* const* d_in, const int* in_sizes, int n_in,
                              void* d_out, int out_size, void* d_ws, size_t ws_size,
                              hipStream_t stream)
{
    (void)in_sizes; (void)n_in; (void)out_size; (void)ws_size;
    const float* x     = (const float*)d_in[0];
    const float* ck    = (const float*)d_in[1];
    const float* cv    = (const float*)d_in[2];
    const float* wqkv  = (const float*)d_in[3];
    const float* bqkv  = (const float*)d_in[4];
    const float* wproj = (const float*)d_in[5];
    const float* bproj = (const float*)d_in[6];
    float* out = (float*)d_out;

    char* ws = (char*)d_ws;
    const size_t szBHSD  = (size_t)BATCH * NH * SEQ * HD * sizeof(__bf16);    // 16 MB
    const size_t szKc    = (size_t)CLEN * NH * HD * sizeof(__bf16);           // 8 MB
    const size_t szAttn  = (size_t)BATCH * SEQ * HID * sizeof(__bf16);        // 16 MB
    const size_t szXb    = (size_t)BATCH * SEQ * HID * sizeof(__bf16);        // 16 MB
    const size_t szWqkvb = (size_t)3 * HID * HID * sizeof(__bf16);            // 24 MB
    const size_t szWprjb = (size_t)HID * HID * sizeof(__bf16);                //  8 MB
    __bf16* Q     = (__bf16*)ws;   ws += szBHSD;
    __bf16* Kn    = (__bf16*)ws;   ws += szBHSD;
    __bf16* Vtn   = (__bf16*)ws;   ws += szBHSD;
    __bf16* Kc    = (__bf16*)ws;   ws += szKc;
    __bf16* Vtc   = (__bf16*)ws;   ws += szKc;
    __bf16* attn  = (__bf16*)ws;   ws += szAttn;
    __bf16* xb    = (__bf16*)ws;   ws += szXb;
    __bf16* wqkvb = (__bf16*)ws;   ws += szWqkvb;
    __bf16* wprjb = (__bf16*)ws;   ws += szWprjb;

    const int nX    = BATCH * SEQ * HID;        //  8,388,608
    const int nWqkv = 3 * HID * HID;            // 12,582,912
    const int nWprj = HID * HID;                //  4,194,304
    const int nKV   = CLEN * NH * HD;           //  4,194,304

    cvt_f32_bf16 <<<(nX    + 255) / 256, 256, 0, stream>>>(x,     xb,    nX);
    cvt_f32_bf16 <<<(nWqkv + 255) / 256, 256, 0, stream>>>(wqkv,  wqkvb, nWqkv);
    cvt_f32_bf16 <<<(nWprj + 255) / 256, 256, 0, stream>>>(wproj, wprjb, nWprj);
    cvt_f32_bf16 <<<(nKV   + 255) / 256, 256, 0, stream>>>(ck,    Kc,    nKV);
    cvt_vcache_t <<<(nKV   + 255) / 256, 256, 0, stream>>>(cv,    Vtc);

    // QKV projection: M=4096, N=6144, K=2048 -> scatter Q/Kn/Vtn (bf16)
    gemm_wmma<1><<<dim3(3 * HID / 128, BATCH * SEQ / 128), 256, 0, stream>>>(
        xb, wqkvb, bqkv, nullptr, Q, Kn, Vtn, 3 * HID);

    // Flash attention over 3072 keys
    flash_attn<<<BATCH * NH * (SEQ / 64), 128, 0, stream>>>(Q, Kc, Kn, Vtc, Vtn, attn);

    // Output projection: M=4096, N=2048, K=2048 -> f32 out + bias
    gemm_wmma<0><<<dim3(HID / 128, BATCH * SEQ / 128), 256, 0, stream>>>(
        attn, wprjb, bproj, out, nullptr, nullptr, nullptr, HID);
}
